// DSSLayer_4964982194340
// MI455X (gfx1250) — matrix operand order repaired and checked by reading, not compiled, not run on hardware
//
#include <hip/hip_runtime.h>

#define L_DIM 4096
#define N_MODES 64
#define EPS_REG 1e-7f
#define KREV_LEN 4256          // 4096 reversed K values + 160 zero pad for causal mask
#define LDS_STRIDE 48          // bf16 elements per LDS row (96 B, multiple of 16 B)
#define UBF_OFF 65536          // ws byte offset of bf16 copy of u

typedef __attribute__((ext_vector_type(16))) __bf16 v16bf;
typedef __attribute__((ext_vector_type(8)))  __bf16 v8bf;
typedef __attribute__((ext_vector_type(8)))  float  v8f;
typedef __attribute__((ext_vector_type(4)))  int    v4i_;

// ---- gfx1250 async global->LDS path (guarded; falls back to sync loads) ----
#if defined(__HIP_DEVICE_COMPILE__) && __has_builtin(__builtin_amdgcn_global_load_async_to_lds_b128)
#define HAS_ASYNC 1
#else
#define HAS_ASYNC 0
#endif

#if HAS_ASYNC
typedef __attribute__((address_space(1))) v4i_ gv4i;   // global 16B vector
typedef __attribute__((address_space(3))) v4i_ lv4i;   // LDS 16B vector
__device__ __forceinline__ void async_cp16(const unsigned short* g, unsigned short* l) {
    __builtin_amdgcn_global_load_async_to_lds_b128((gv4i*)g, (lv4i*)l, 0, 0);
}
#if __has_builtin(__builtin_amdgcn_s_wait_asynccnt)
#define WAIT_ASYNC(n) __builtin_amdgcn_s_wait_asynccnt(n)
#else
#define WAIT_ASYNC(n) asm volatile("s_wait_asynccnt " #n ::: "memory")
#endif
#else
#define WAIT_ASYNC(n) ((void)0)
#endif

__device__ __forceinline__ unsigned short f32_to_bf16_bits(float f) {
    union { float f; unsigned int u; } v; v.f = f;
    unsigned int u = v.u;
    unsigned int r = u + 0x7FFFu + ((u >> 16) & 1u);   // round-to-nearest-even
    return (unsigned short)(r >> 16);
}

__device__ __forceinline__ __bf16 bits_to_bf16(unsigned short s) {
    union { unsigned short s; __bf16 b; } v; v.s = s; return v.b;
}

// ---------------------------------------------------------------------------
// Kernel 1: per-mode complex-softmax stats.
// z_n = step*Lambda_n ; m_n = argmax_l Re(z_n*l) (= 0 or L-1 analytically).
// T_n = sum_l exp(z_n*(l-m_n)) ; c_n = (W_n/Lambda_n) * conj(T_n)/(|T_n|^2+eps)
// ---------------------------------------------------------------------------
__global__ void dss_row_stats(const float* __restrict__ Lre,
                              const float* __restrict__ Lim,
                              const float* __restrict__ W,
                              const float* __restrict__ log_step,
                              float* __restrict__ cRe, float* __restrict__ cIm,
                              float* __restrict__ zr_out, float* __restrict__ zi_out) {
    const int n   = blockIdx.x;
    const int tid = threadIdx.x;
    const float step = expf(log_step[0]);
    const float lr = Lre[n], li = Lim[n];
    const float zr = step * lr, zi = step * li;
    const float m  = (zr > 0.0f) ? (float)(L_DIM - 1) : 0.0f;

    float sr = 0.0f, si = 0.0f;
    for (int l = tid; l < L_DIM; l += 256) {
        float d  = (float)l - m;
        float er = expf(zr * d);
        float s, c;
        sincosf(zi * d, &s, &c);
        sr += er * c;
        si += er * s;
    }
    __shared__ float red[2][256];
    red[0][tid] = sr; red[1][tid] = si;
    __syncthreads();
    for (int off = 128; off > 0; off >>= 1) {
        if (tid < off) {
            red[0][tid] += red[0][tid + off];
            red[1][tid] += red[1][tid + off];
        }
        __syncthreads();
    }
    if (tid == 0) {
        sr = red[0][0]; si = red[1][0];
        float denom = sr * sr + si * si + EPS_REG;
        float rr =  sr / denom;
        float ri = -si / denom;
        float wr = W[2 * n], wi = W[2 * n + 1];
        float ld = lr * lr + li * li;
        float wpr = (wr * lr + wi * li) / ld;     // W / Lambda
        float wpi = (wi * lr - wr * li) / ld;
        cRe[n] = wpr * rr - wpi * ri;
        cIm[n] = wpr * ri + wpi * rr;
        zr_out[n] = zr; zi_out[n] = zi;
    }
}

// ---------------------------------------------------------------------------
// Kernel 2: K[l] = sum_n Re(c_n * exp(z_n*(l - m_n))), stored REVERSED in bf16
// with 160 zero-pad entries. D is folded into K[0] (the s==t diagonal term),
// so the GEMM epilogue needs no u reload.
// ---------------------------------------------------------------------------
__global__ void dss_build_k(const float* __restrict__ cRe, const float* __restrict__ cIm,
                            const float* __restrict__ zr,  const float* __restrict__ zi,
                            const float* __restrict__ D,
                            unsigned short* __restrict__ Krev) {
    int gid = blockIdx.x * 256 + threadIdx.x;
    if (gid >= KREV_LEN) return;
    if (gid >= L_DIM) { Krev[gid] = 0; return; }
    const int l = gid;
    float acc = (l == 0) ? D[0] : 0.0f;   // y += D*u  ==  K[0] += D
    #pragma unroll 4
    for (int n = 0; n < N_MODES; ++n) {
        float zrn = zr[n];
        float m   = (zrn > 0.0f) ? (float)(L_DIM - 1) : 0.0f;
        float d   = (float)l - m;
        float er  = expf(zrn * d);
        float s, c;
        sincosf(zi[n] * d, &s, &c);
        acc += cRe[n] * (er * c) - cIm[n] * (er * s);
    }
    Krev[(L_DIM - 1) - l] = f32_to_bf16_bits(acc);
}

// ---------------------------------------------------------------------------
// Kernel 3: one-time f32 -> bf16 conversion of u into workspace.
// ---------------------------------------------------------------------------
__global__ void dss_convert_u(const float* __restrict__ u, unsigned short* __restrict__ ubf) {
    size_t base = ((size_t)blockIdx.x * 256 + threadIdx.x) * 8;
    float4 a = *(const float4*)(u + base);
    float4 b = *(const float4*)(u + base + 4);
    uint4 o;
    o.x = (unsigned)f32_to_bf16_bits(a.x) | ((unsigned)f32_to_bf16_bits(a.y) << 16);
    o.y = (unsigned)f32_to_bf16_bits(a.z) | ((unsigned)f32_to_bf16_bits(a.w) << 16);
    o.z = (unsigned)f32_to_bf16_bits(b.x) | ((unsigned)f32_to_bf16_bits(b.y) << 16);
    o.w = (unsigned)f32_to_bf16_bits(b.z) | ((unsigned)f32_to_bf16_bits(b.w) << 16);
    *(uint4*)(ubf + base) = o;
}

// ---------------------------------------------------------------------------
// Kernel 4 (main path): causal Toeplitz GEMM, bf16 WMMA, double-buffered LDS
// tiles filled by GLOBAL_LOAD_ASYNC_TO_LDS_B128.
// Block: 256 threads (8 waves), 128x128 output tile, k-step 32.
// ---------------------------------------------------------------------------
__global__ __launch_bounds__(256)
void dss_gemm_bf16(const unsigned short* __restrict__ ubf,
                   const unsigned short* __restrict__ Krev,
                   float* __restrict__ out) {
    __shared__ __attribute__((aligned(16))) unsigned short ldsU[2][128 * LDS_STRIDE]; // 24 KB
    __shared__ __attribute__((aligned(16))) unsigned short ldsK[KREV_LEN];            // 8.5 KB

    const int tid     = threadIdx.x;
    const int lane    = tid & 31;
    const int laneLo  = lane & 15;
    const int h       = lane >> 4;
    const int wid     = tid >> 5;
    const int waveM   = wid >> 2;        // 0..1 -> 64-row half
    const int waveN   = wid & 3;         // 0..3 -> 32-col quarter
    const int colBase = blockIdx.x * 128;
    const int rowBase = blockIdx.y * 128;

    for (int i = tid; i < KREV_LEN; i += 256) ldsK[i] = Krev[i];

    const int ldRow  = tid >> 1;         // 0..127
    const int ldHalf = tid & 1;          // 16 bf16 each

    const unsigned short* gRow = ubf + (size_t)(rowBase + ldRow) * L_DIM + ldHalf * 16;

    const v8f vzero = {0.f, 0.f, 0.f, 0.f, 0.f, 0.f, 0.f, 0.f};
    v8f acc[4][2];
    #pragma unroll
    for (int mt = 0; mt < 4; ++mt)
        #pragma unroll
        for (int nt = 0; nt < 2; ++nt) acc[mt][nt] = vzero;

    const int kBlocks = blockIdx.x * 4 + 4;   // causality: s0 <= t0+127

    // ---- stage tile kb into LDS buffer buf (this thread's 32 bytes) ----
    auto issue_tile = [&](int kb, int buf) {
        const unsigned short* g = gRow + kb * 32;
        unsigned short* l = &ldsU[buf][ldRow * LDS_STRIDE + ldHalf * 16];
#if HAS_ASYNC
        async_cp16(g, l);
        async_cp16(g + 8, l + 8);
#else
        uint4 d0 = *(const uint4*)(g);
        uint4 d1 = *(const uint4*)(g + 8);
        *(uint4*)(l)     = d0;
        *(uint4*)(l + 8) = d1;
#endif
    };

    issue_tile(0, 0);

    for (int kb = 0; kb < kBlocks; ++kb) {
        const int s0  = kb * 32;
        const int buf = kb & 1;
        if (kb + 1 < kBlocks) {
            issue_tile(kb + 1, buf ^ 1);   // prefetch next tile into other buffer
            WAIT_ASYNC(2);                 // this thread's tile-kb copies done
        } else {
            WAIT_ASYNC(0);
        }
        __syncthreads();                   // tile kb visible to all waves

        // B fragments from reversed K (lane = column t, half h = K rows 16h..16h+15)
        v16bf bfrag[2];
        #pragma unroll
        for (int nt = 0; nt < 2; ++nt) {
            const int t    = colBase + waveN * 32 + nt * 16 + laneLo;
            const int base = (L_DIM - 1) - t + s0 + h * 16;   // in [0, KREV_LEN)
            v16bf bfv;
            #pragma unroll
            for (int e = 0; e < 16; ++e) bfv[e] = bits_to_bf16(ldsK[base + e]);
            bfrag[nt] = bfv;
        }

        // A fragments (elems 0..15 <-> K=16h..16h+15) + WMMA
        #pragma unroll
        for (int mt = 0; mt < 4; ++mt) {
            const int rowL = waveM * 64 + mt * 16 + laneLo;
            const unsigned short* ap = &ldsU[buf][rowL * LDS_STRIDE + h * 16];
            union { v16bf v; v8bf hv[2]; } af;
            af.hv[0] = *(const v8bf*)(ap);
            af.hv[1] = *(const v8bf*)(ap + 8);
            #pragma unroll
            for (int nt = 0; nt < 2; ++nt) {
                acc[mt][nt] = __builtin_amdgcn_wmma_f32_16x16x32_bf16(
                    false, af.v, false, bfrag[nt],
                    (short)0, acc[mt][nt], false, false);
            }
        }
        __syncthreads();                   // done reading buf before it is refilled
    }

    // ---- epilogue: pure coalesced stores (D already folded into K) ----
    #pragma unroll
    for (int mt = 0; mt < 4; ++mt) {
        #pragma unroll
        for (int nt = 0; nt < 2; ++nt) {
            const int t = colBase + waveN * 32 + nt * 16 + laneLo;
            #pragma unroll
            for (int r = 0; r < 8; ++r) {
                const int row = rowBase + waveM * 64 + mt * 16 + h * 8 + r;
                out[(size_t)row * L_DIM + t] = acc[mt][nt][r];
            }
        }
    }
}

// ---------------------------------------------------------------------------
// Kernel 4b (fallback if ws too small): same GEMM, f32 u loaded + converted
// on the fly (round-1 scheme), D folded into K so epilogue is stores only.
// ---------------------------------------------------------------------------
__global__ __launch_bounds__(256)
void dss_gemm_f32(const float* __restrict__ u, const unsigned short* __restrict__ Krev,
                  float* __restrict__ out) {
    __shared__ __attribute__((aligned(16))) __bf16 ldsU[128 * LDS_STRIDE];
    __shared__ unsigned short ldsK[KREV_LEN];

    const int tid     = threadIdx.x;
    const int lane    = tid & 31;
    const int laneLo  = lane & 15;
    const int h       = lane >> 4;
    const int wid     = tid >> 5;
    const int waveM   = wid >> 2;
    const int waveN   = wid & 3;
    const int colBase = blockIdx.x * 128;
    const int rowBase = blockIdx.y * 128;

    for (int i = tid; i < KREV_LEN; i += 256) ldsK[i] = Krev[i];

    const v8f vzero = {0.f, 0.f, 0.f, 0.f, 0.f, 0.f, 0.f, 0.f};
    v8f acc[4][2];
    #pragma unroll
    for (int mt = 0; mt < 4; ++mt)
        #pragma unroll
        for (int nt = 0; nt < 2; ++nt) acc[mt][nt] = vzero;

    const int kBlocks = blockIdx.x * 4 + 4;
    const int ldRow  = tid >> 1;
    const int ldHalf = tid & 1;

    for (int kb = 0; kb < kBlocks; ++kb) {
        const int s0 = kb * 32;
        {
            const float* src = u + (size_t)(rowBase + ldRow) * L_DIM + s0 + ldHalf * 16;
            float4 f0 = *(const float4*)(src);
            float4 f1 = *(const float4*)(src + 4);
            float4 f2 = *(const float4*)(src + 8);
            float4 f3 = *(const float4*)(src + 12);
            if (kb + 1 < kBlocks) __builtin_prefetch(src + 32, 0, 0);
            v8bf b0 = { (__bf16)f0.x, (__bf16)f0.y, (__bf16)f0.z, (__bf16)f0.w,
                        (__bf16)f1.x, (__bf16)f1.y, (__bf16)f1.z, (__bf16)f1.w };
            v8bf b1 = { (__bf16)f2.x, (__bf16)f2.y, (__bf16)f2.z, (__bf16)f2.w,
                        (__bf16)f3.x, (__bf16)f3.y, (__bf16)f3.z, (__bf16)f3.w };
            __bf16* dst = &ldsU[ldRow * LDS_STRIDE + ldHalf * 16];
            *(v8bf*)(dst)     = b0;
            *(v8bf*)(dst + 8) = b1;
        }
        __syncthreads();

        v16bf bfrag[2];
        #pragma unroll
        for (int nt = 0; nt < 2; ++nt) {
            const int t    = colBase + waveN * 32 + nt * 16 + laneLo;
            const int base = (L_DIM - 1) - t + s0 + h * 16;
            v16bf bfv;
            #pragma unroll
            for (int e = 0; e < 16; ++e) bfv[e] = bits_to_bf16(ldsK[base + e]);
            bfrag[nt] = bfv;
        }
        #pragma unroll
        for (int mt = 0; mt < 4; ++mt) {
            const int rowL = waveM * 64 + mt * 16 + laneLo;
            const __bf16* ap = &ldsU[rowL * LDS_STRIDE + h * 16];
            union { v16bf v; v8bf hv[2]; } af;
            af.hv[0] = *(const v8bf*)(ap);
            af.hv[1] = *(const v8bf*)(ap + 8);
            #pragma unroll
            for (int nt = 0; nt < 2; ++nt) {
                acc[mt][nt] = __builtin_amdgcn_wmma_f32_16x16x32_bf16(
                    false, af.v, false, bfrag[nt],
                    (short)0, acc[mt][nt], false, false);
            }
        }
        __syncthreads();
    }

    #pragma unroll
    for (int mt = 0; mt < 4; ++mt) {
        #pragma unroll
        for (int nt = 0; nt < 2; ++nt) {
            const int t = colBase + waveN * 32 + nt * 16 + laneLo;
            #pragma unroll
            for (int r = 0; r < 8; ++r) {
                const int row = rowBase + waveM * 64 + mt * 16 + h * 8 + r;
                out[(size_t)row * L_DIM + t] = acc[mt][nt][r];
            }
        }
    }
}

// ---------------------------------------------------------------------------
extern "C" void kernel_launch(void* const* d_in, const int* in_sizes, int n_in,
                              void* d_out, int out_size, void* d_ws, size_t ws_size,
                              hipStream_t stream) {
    const float* u        = (const float*)d_in[0];   // (4096, 4096)
    const float* Lre      = (const float*)d_in[1];   // (64,)
    const float* Lim      = (const float*)d_in[2];   // (64,)
    const float* W        = (const float*)d_in[3];   // (1, 64, 2)
    const float* D        = (const float*)d_in[4];   // (1,)
    const float* log_step = (const float*)d_in[5];   // (1,)
    float* out = (float*)d_out;

    char* ws = (char*)d_ws;
    float* cRe = (float*)(ws);
    float* cIm = (float*)(ws + 256);
    float* zr  = (float*)(ws + 512);
    float* zi  = (float*)(ws + 768);
    unsigned short* Krev = (unsigned short*)(ws + 4096);
    unsigned short* ubf  = (unsigned short*)(ws + UBF_OFF);

    dss_row_stats<<<N_MODES, 256, 0, stream>>>(Lre, Lim, W, log_step, cRe, cIm, zr, zi);
    dss_build_k<<<(KREV_LEN + 255) / 256, 256, 0, stream>>>(cRe, cIm, zr, zi, D, Krev);

    dim3 grid(L_DIM / 128, L_DIM / 128);
    const size_t need = (size_t)UBF_OFF + sizeof(unsigned short) * (size_t)L_DIM * L_DIM;
    if (ws_size >= need) {
        dss_convert_u<<<(L_DIM * (L_DIM / 8)) / 256, 256, 0, stream>>>(u, ubf);
        dss_gemm_bf16<<<grid, 256, 0, stream>>>(ubf, Krev, out);
    } else {
        dss_gemm_f32<<<grid, 256, 0, stream>>>(u, Krev, out);
    }
}
